// VallinaDecoder_73675868995758
// MI455X (gfx1250) — matrix operand order repaired and checked by reading, compiled, not run on hardware
//
#include <hip/hip_runtime.h>
#include <hip/hip_bf16.h>
#include <math.h>

// Problem dims (compile-time constants from the reference)
#define B_  64
#define S_  128
#define E_  512
#define H_  1024
#define C_  1024
#define A_  512
#define D2_ 1024
#define V_  32000

typedef __attribute__((ext_vector_type(16))) __bf16 v16bf;
typedef __attribute__((ext_vector_type(8)))  float  v8f;

// Native f32 -> bf16 (clang lowers pairs to v_cvt_pk_bf16_f32 on gfx1250)
__device__ __forceinline__ v16bf cvt16(const float* __restrict__ f) {
    v16bf o;
#pragma unroll
    for (int i = 0; i < 16; ++i) o[i] = (__bf16)f[i];
    return o;
}

// A 16x32 bf16 fragment (per CDNA5 ISA 7.12.2):
// elem i<8 -> k = 8*half + i ; elem i>=8 -> k = 8 + 8*half + i
// caller passes ap = &A[row][kk] + 8*half
__device__ __forceinline__ v16bf load_a_frag(const float* __restrict__ ap) {
    float fa[16];
    *(float4*)(fa + 0)  = *(const float4*)(ap + 0);
    *(float4*)(fa + 4)  = *(const float4*)(ap + 4);
    *(float4*)(fa + 8)  = *(const float4*)(ap + 16);
    *(float4*)(fa + 12) = *(const float4*)(ap + 20);
    return cvt16(fa);
}

// B 32x16 bf16 fragment: elem i<8 -> k = 8*half + i ; elem i>=8 -> k = 8 + 8*half + i (col = lane&15)
// caller passes bp = &W[col][kk] + 8*half
__device__ __forceinline__ v16bf load_b_frag(const float* __restrict__ bp) {
    float fb[16];
    *(float4*)(fb + 0)  = *(const float4*)(bp + 0);
    *(float4*)(fb + 4)  = *(const float4*)(bp + 4);
    *(float4*)(fb + 8)  = *(const float4*)(bp + 16);
    *(float4*)(fb + 12) = *(const float4*)(bp + 20);
    return cvt16(fb);
}

// ---------------------------------------------------------------------------
// Generic GEMM: C[M,N] = act( A[M,K] @ W[N,K]^T + bias [+ C] )
// fp32 in/out, in-register bf16 conversion, v_wmma_f32_16x16x32_bf16, f32 acc.
// Each wave computes a 64x32 block (4 M-tiles x 2 N-tiles): per K-step of 32
// we convert 6 fragments and issue 8 WMMAs, so weight streaming stays the
// bottleneck (~230 MB total -> ~10 us at 23.3 TB/s). M=64 strips mean each
// weight element is fetched from DRAM exactly once.
// Requires M % 64 == 0, N % 32 == 0, K % 32 == 0 (true for all call sites).
// ---------------------------------------------------------------------------
template<bool ACCUM, bool TANH>
__global__ __launch_bounds__(256)
void gemm_wmma(const float* __restrict__ Aa, int lda,
               const float* __restrict__ Bw, int ldb,
               const float* __restrict__ bias,
               float* __restrict__ Cc, int ldc,
               int M, int N, int K) {
    const int lane = threadIdx.x & 31;
    const int wv   = threadIdx.x >> 5;
    const int Mb = M >> 6, Nb = N >> 5;
    const long nwaves = (long)Mb * Nb;
    const long wt = (long)blockIdx.x * 8 + wv;
    if (wt >= nwaves) return;            // whole wave exits; EXEC stays all-1s
    const int mb   = (int)(wt % Mb);
    const int nb   = (int)(wt / Mb);
    const int half = lane >> 4;
    const int r    = lane & 15;

    const float* Ar[4];
#pragma unroll
    for (int mt = 0; mt < 4; ++mt)
        Ar[mt] = Aa + (size_t)(mb * 64 + mt * 16 + r) * lda + 8 * half;
    const float* Br[2];
#pragma unroll
    for (int nt = 0; nt < 2; ++nt)
        Br[nt] = Bw + (size_t)(nb * 32 + nt * 16 + r) * ldb + 8 * half;

    v8f acc[4][2] = {};
    for (int kk = 0; kk < K; kk += 32) {
        v16bf af[4];
#pragma unroll
        for (int mt = 0; mt < 4; ++mt) af[mt] = load_a_frag(Ar[mt] + kk);
#pragma unroll
        for (int nt = 0; nt < 2; ++nt) {
            v16bf bf = load_b_frag(Br[nt] + kk);
#pragma unroll
            for (int mt = 0; mt < 4; ++mt)
                acc[mt][nt] = __builtin_amdgcn_wmma_f32_16x16x32_bf16(
                    false, af[mt], false, bf, (short)0, acc[mt][nt], false, false);
        }
    }

    // C 16x16 f32 layout: VGPR j -> row = j + 8*half, col = lane&15
    float bn[2];
#pragma unroll
    for (int nt = 0; nt < 2; ++nt)
        bn[nt] = bias ? bias[nb * 32 + nt * 16 + r] : 0.0f;
#pragma unroll
    for (int mt = 0; mt < 4; ++mt)
#pragma unroll
        for (int nt = 0; nt < 2; ++nt)
#pragma unroll
            for (int j = 0; j < 8; ++j) {
                size_t idx = (size_t)(mb * 64 + mt * 16 + j + 8 * half) * ldc
                           + nb * 32 + nt * 16 + r;
                float v = acc[mt][nt][j] + bn[nt];
                if (ACCUM) v += Cc[idx];
                if (TANH)  v = tanhf(v);
                Cc[idx] = v;
            }
}

// ---------------------------------------------------------------------------
// GRU gate fusion: h' = (1-z)*n + z*h  (PyTorch GRUCell semantics)
// ---------------------------------------------------------------------------
__global__ void gru_gate(const float* __restrict__ gi, const float* __restrict__ gh,
                         const float* __restrict__ h, float* __restrict__ hout) {
    int tid = blockIdx.x * blockDim.x + threadIdx.x;
    if (tid >= B_ * H_) return;
    int b = tid / H_, j = tid % H_;
    size_t base = (size_t)b * 3 * H_ + j;
    float ir = gi[base], iz = gi[base + H_], in_ = gi[base + 2 * H_];
    float hr = gh[base], hz = gh[base + H_], hn  = gh[base + 2 * H_];
    float rg = 1.0f / (1.0f + expf(-(ir + hr)));
    float zg = 1.0f / (1.0f + expf(-(iz + hz)));
    float ng = tanhf(in_ + rg * hn);
    hout[tid] = (1.0f - zg) * ng + zg * h[tid];
}

// ---------------------------------------------------------------------------
// logit[b,s] = sum_a tanh(attn_h[b,s,a] + q[b,a]) * Wa[a] + ba ; mask -> -inf
// One wave32 per (b,s) row; shfl_xor reduction.
// ---------------------------------------------------------------------------
__global__ __launch_bounds__(256)
void attn_logits(const float* __restrict__ ah, const float* __restrict__ q,
                 const float* __restrict__ Wa, const float* __restrict__ ba,
                 const unsigned char* __restrict__ mask, float* __restrict__ logit) {
    int row = blockIdx.x * 8 + (threadIdx.x >> 5);
    if (row >= B_ * S_) return;
    int lane = threadIdx.x & 31;
    int b = row / S_;
    float sum = 0.0f;
    for (int a = lane; a < A_; a += 32)
        sum += tanhf(ah[(size_t)row * A_ + a] + q[(size_t)b * A_ + a]) * Wa[a];
#pragma unroll
    for (int off = 16; off > 0; off >>= 1) sum += __shfl_xor(sum, off, 32);
    if (lane == 0)
        logit[row] = mask[row] ? (sum + ba[0]) : -__builtin_inff();
}

// Softmax over S=128 per batch row; one wave, 4 elems/lane.
__global__ void softmax128(const float* __restrict__ logit, float* __restrict__ w) {
    int b = blockIdx.x, lane = threadIdx.x;
    float v[4];
#pragma unroll
    for (int j = 0; j < 4; ++j) v[j] = logit[b * S_ + lane + 32 * j];
    float mx = fmaxf(fmaxf(v[0], v[1]), fmaxf(v[2], v[3]));
#pragma unroll
    for (int off = 16; off > 0; off >>= 1) mx = fmaxf(mx, __shfl_xor(mx, off, 32));
    float e[4], s = 0.0f;
#pragma unroll
    for (int j = 0; j < 4; ++j) { e[j] = expf(v[j] - mx); s += e[j]; }
#pragma unroll
    for (int off = 16; off > 0; off >>= 1) s += __shfl_xor(s, off, 32);
    float inv = 1.0f / s;
#pragma unroll
    for (int j = 0; j < 4; ++j) w[b * S_ + lane + 32 * j] = e[j] * inv;
}

// ctx[b,c] = sum_s w[b,s] * enc[b,s,c]   (coalesced over c; weights in LDS)
__global__ void attn_context(const float* __restrict__ w, const float* __restrict__ enc,
                             float* __restrict__ ctx) {
    __shared__ float wl[S_];
    int b = blockIdx.y;
    int c = blockIdx.x * 256 + threadIdx.x;
    if (threadIdx.x < S_) wl[threadIdx.x] = w[b * S_ + threadIdx.x];
    __syncthreads();
    float sum = 0.0f;
    const float* e = enc + (size_t)b * S_ * C_ + c;
#pragma unroll 4
    for (int s = 0; s < S_; ++s) sum += wl[s] * e[(size_t)s * C_];
    ctx[(size_t)b * C_ + c] = sum;
}

// ---------------------------------------------------------------------------
static void gemm(const float* A, int lda, const float* W, int ldb, const float* bias,
                 float* C, int ldc, int M, int N, int K, bool accum, bool act,
                 hipStream_t s) {
    long waves = (long)(M / 64) * (N / 32);
    dim3 grid((unsigned)((waves + 7) / 8));
    if (!accum && !act)      gemm_wmma<false, false><<<grid, 256, 0, s>>>(A, lda, W, ldb, bias, C, ldc, M, N, K);
    else if (accum && !act)  gemm_wmma<true,  false><<<grid, 256, 0, s>>>(A, lda, W, ldb, bias, C, ldc, M, N, K);
    else if (accum && act)   gemm_wmma<true,  true ><<<grid, 256, 0, s>>>(A, lda, W, ldb, bias, C, ldc, M, N, K);
    else                     gemm_wmma<false, true ><<<grid, 256, 0, s>>>(A, lda, W, ldb, bias, C, ldc, M, N, K);
}

extern "C" void kernel_launch(void* const* d_in, const int* in_sizes, int n_in,
                              void* d_out, int out_size, void* d_ws, size_t ws_size,
                              hipStream_t stream) {
    const float* x      = (const float*)d_in[0];
    const float* hidden = (const float*)d_in[1];
    const unsigned char* enc_mask = (const unsigned char*)d_in[2];
    const float* enc    = (const float*)d_in[3];
    const float* g1Wih  = (const float*)d_in[4];
    const float* g1Whh  = (const float*)d_in[5];
    const float* g1bih  = (const float*)d_in[6];
    const float* g1bhh  = (const float*)d_in[7];
    const float* g2Wih  = (const float*)d_in[8];
    const float* g2Whh  = (const float*)d_in[9];
    const float* g2bih  = (const float*)d_in[10];
    const float* g2bhh  = (const float*)d_in[11];
    const float* Wq  = (const float*)d_in[12];
    const float* bq  = (const float*)d_in[13];
    const float* Wv  = (const float*)d_in[14];
    const float* bvv = (const float*)d_in[15];
    const float* Wa  = (const float*)d_in[16];
    const float* ba  = (const float*)d_in[17];
    const float* e2oW = (const float*)d_in[18];
    const float* e2ob = (const float*)d_in[19];
    const float* h2oW = (const float*)d_in[20];
    const float* h2ob = (const float*)d_in[21];
    const float* c2oW = (const float*)d_in[22];
    const float* c2ob = (const float*)d_in[23];
    const float* affW = (const float*)d_in[24];
    const float* affb = (const float*)d_in[25];

    // Workspace carve-up (floats), ~21 MB total
    float* ws  = (float*)d_ws;
    float* gi1 = ws;                          // 64*3072
    float* gh1 = gi1 + B_ * 3 * H_;           // 64*3072
    float* h1  = gh1 + B_ * 3 * H_;           // 64*1024
    float* q   = h1  + B_ * H_;               // 64*512
    float* ah  = q   + B_ * A_;               // 8192*512
    float* lg  = ah  + (size_t)B_ * S_ * A_;  // 8192
    float* wts = lg  + B_ * S_;               // 8192
    float* ctx = wts + B_ * S_;               // 64*1024
    float* gi2 = ctx + B_ * C_;               // 64*3072
    float* gh2 = gi2 + B_ * 3 * H_;           // 64*3072
    float* t   = gh2 + B_ * 3 * H_;           // 64*1024

    float* out = (float*)d_out;               // [64,32000]
    float* h2  = out + (size_t)B_ * V_;       // [64,1024] (tuple output #2)

    // --- GRU cell 1 ---
    gemm(x,      E_, g1Wih, E_, g1bih, gi1, 3 * H_, B_, 3 * H_, E_, false, false, stream);
    gemm(hidden, H_, g1Whh, H_, g1bhh, gh1, 3 * H_, B_, 3 * H_, H_, false, false, stream);
    gru_gate<<<(B_ * H_ + 255) / 256, 256, 0, stream>>>(gi1, gh1, hidden, h1);

    // --- Additive attention ---
    gemm(enc, C_, Wv, C_, bvv, ah, A_, B_ * S_, A_, C_, false, false, stream);  // [8192,512]
    gemm(h1,  H_, Wq, H_, bq,  q,  A_, B_,      A_, H_, false, false, stream);  // [64,512]
    attn_logits<<<(B_ * S_ + 7) / 8, 256, 0, stream>>>(ah, q, Wa, ba, enc_mask, lg);
    softmax128<<<B_, 32, 0, stream>>>(lg, wts);
    attn_context<<<dim3(C_ / 256, B_), 256, 0, stream>>>(wts, enc, ctx);

    // --- GRU cell 2 ---
    gemm(ctx, C_, g2Wih, C_, g2bih, gi2, 3 * H_, B_, 3 * H_, C_, false, false, stream);
    gemm(h1,  H_, g2Whh, H_, g2bhh, gh2, 3 * H_, B_, 3 * H_, H_, false, false, stream);
    gru_gate<<<(B_ * H_ + 255) / 256, 256, 0, stream>>>(gi2, gh2, h1, h2);

    // --- Fused readout: t = tanh(x@e2o^T + h2@h2o^T + ctx@c2o^T + biases) ---
    gemm(x,   E_,  e2oW, E_,  e2ob, t, D2_, B_, D2_, E_,  false, false, stream);
    gemm(h2,  H_,  h2oW, H_,  h2ob, t, D2_, B_, D2_, H_,  true,  false, stream);
    gemm(ctx, C_,  c2oW, C_,  c2ob, t, D2_, B_, D2_, C_,  true,  true,  stream);

    // --- Final affine: out = t @ aff_W^T + aff_b  (131 MB weight stream) ---
    gemm(t, D2_, affW, D2_, affb, out, V_, B_, V_, D2_, false, false, stream);
}
